// SymSimGCNNet_31207232372904
// MI455X (gfx1250) — matrix operand order repaired and checked by reading, not compiled, not run on hardware
//
#include <hip/hip_runtime.h>
#include <hip/hip_bf16.h>

// SymSimGCNNet on MI455X (gfx1250, wave32, WMMA).
//
// Key insight: every graph in the batch shares the SAME fully-connected
// 62-node topology and the SAME learnable weight matrix, so the normalized
// adjacency Ahat = D^-1/2 W D^-1/2 (and its square A2 = Ahat^2, covering the
// K=2 propagation hops) is ONE 62x62 matrix. The whole network collapses to
// per-graph dense matmuls:
//   h1 = relu(A2 @ (X @ W1^T) + b1)        (62x5 -> 62x64)
//   h2 = relu((A2 @ h1) @ W2^T + b2)       (two 64x64x64 matmuls)
//   out = (colsum_{rows<62} h2) @ Wfc^T + bfc
// The three 64^3 matmuls per graph run on v_wmma_f32_16x16x32_f16.

#define N_NODES 62
#define IN_CH   5
#define HID     64
#define OUT_CH  3
#define BATCH   1024
#define PITCH   72   // f16 row pitch in LDS/ws (72*2B=144B -> conflict-free)

typedef _Float16 v16h __attribute__((ext_vector_type(16)));
typedef _Float16 v8h  __attribute__((ext_vector_type(8)));
typedef float    v8f  __attribute__((ext_vector_type(8)));

static __device__ inline v8f zero8() {
    v8f z = {0.f, 0.f, 0.f, 0.f, 0.f, 0.f, 0.f, 0.f};
    return z;
}

// ---- WMMA fragment loaders (layouts per CDNA5 ISA 7.12.2, wave32) ----------
// A (16x32 f16): lanes 0-15 hold row M=lane, elems 0..7 = K[kk+0..7],
//                elems 8..15 = K[kk+16..23]; lanes 16-31 same row, K +8.
static __device__ inline v16h load_afrag(const _Float16* A, int tm, int kk,
                                         int hi, int ln) {
    const _Float16* p = A + (tm * 16 + ln) * PITCH + kk + hi * 8;
    v16h a;
    *(v8h*)&a       = *(const v8h*)p;         // K kbase .. +7
    *((v8h*)&a + 1) = *(const v8h*)(p + 16);  // K kbase+16 .. +23
    return a;
}
// B (32x16 f16) read from a TRANSPOSED (N-major) buffer Bt[col][k]:
// lanes 0-15 hold col N=lane, elems 0..7 = K[kk+0..7], elems 8..15 = K[kk+16..23]; lanes 16-31 same col, K +8.
static __device__ inline v16h load_bfrag(const _Float16* Bt, int tn, int kk,
                                         int hi, int ln) {
    const _Float16* p = Bt + (tn * 16 + ln) * PITCH + kk + hi * 8;
    v16h b;
    *(v8h*)&b       = *(const v8h*)p;        // K kbase .. +7
    *((v8h*)&b + 1) = *(const v8h*)(p + 16); // K kbase+16 .. +23
    return b;
}

// One 64x64x64 matmul step: each of 8 waves owns two 16x16 output tiles.
static __device__ inline void mm64(const _Float16* A, const _Float16* Bt,
                                   int wave, int hi, int ln, v8f acc[2]) {
#pragma unroll
    for (int lt = 0; lt < 2; ++lt) {
        const int t = wave * 2 + lt, tm = t >> 2, tn = t & 3;
#pragma unroll
        for (int kk = 0; kk < 64; kk += 32) {
            v16h a = load_afrag(A, tm, kk, hi, ln);
            v16h b = load_bfrag(Bt, tn, kk, hi, ln);
            acc[lt] = __builtin_amdgcn_wmma_f32_16x16x32_f16(
                false, a, false, b, (short)0, acc[lt], false, false);
        }
    }
}

// ---- Prep: build Ahat from tril vector, square it, emit f16 A2 & W2 --------
__global__ __launch_bounds__(256) void gcn_prep_kernel(
    const float* __restrict__ tril, const float* __restrict__ w2,
    _Float16* __restrict__ a2g, _Float16* __restrict__ w2g) {
    const int tid = threadIdx.x;
    if (blockIdx.x == 1) {  // f16 copy of W2 (acts as Bt for h @ W2^T)
        for (int idx = tid; idx < HID * HID; idx += 256)
            w2g[(idx >> 6) * PITCH + (idx & 63)] = (_Float16)w2[idx];
        return;
    }
    __shared__ float M[64 * 64];
    __shared__ float dinv[64];
    // symmetric dense W from tril vector (diagonal kept once), zero-padded
    for (int idx = tid; idx < 64 * 64; idx += 256) {
        const int i = idx >> 6, j = idx & 63;
        float v = 0.f;
        if (i < N_NODES && j < N_NODES)
            v = (i >= j) ? tril[i * (i + 1) / 2 + j] : tril[j * (j + 1) / 2 + i];
        M[idx] = v;
    }
    __syncthreads();
    if (tid < 64) {
        float deg = 0.f;
        for (int j = 0; j < 64; ++j) deg += fabsf(M[tid * 64 + j]);
        dinv[tid] = (deg > 0.f) ? rsqrtf(deg) : 0.f;
    }
    __syncthreads();
    for (int idx = tid; idx < 64 * 64; idx += 256)  // Ahat in place
        M[idx] *= dinv[idx >> 6] * dinv[idx & 63];
    __syncthreads();
    for (int idx = tid; idx < 64 * 64; idx += 256) {  // A2 = Ahat^2 -> f16
        const int i = idx >> 6, j = idx & 63;
        float s = 0.f;
        for (int k = 0; k < 64; ++k) s += M[i * 64 + k] * M[k * 64 + j];
        a2g[i * PITCH + j] = (_Float16)s;
    }
}

// ---- Main: one workgroup (8 wave32) per graph ------------------------------
__global__ __launch_bounds__(256) void gcn_graph_kernel(
    const float* __restrict__ x, const float* __restrict__ w1,
    const float* __restrict__ b1, const float* __restrict__ b2,
    const float* __restrict__ wfc, const float* __restrict__ bfc,
    const _Float16* __restrict__ a2g, const _Float16* __restrict__ w2g,
    float* __restrict__ out) {
    __shared__ __align__(16) _Float16 sA2[64 * PITCH];  // A2, row-major
    __shared__ __align__(16) _Float16 sB[64 * PITCH];   // Tt, then H1t (N-major)
    __shared__ __align__(16) _Float16 sU[64 * PITCH];   // U = A2 @ h1, row-major
    __shared__ __align__(16) _Float16 sW2[64 * PITCH];  // W2 (= (W2^T)^T)
    __shared__ float sX[N_NODES * IN_CH + 2];
    __shared__ float sW1[HID * IN_CH];
    __shared__ float sb1[64], sb2[64], sPool[64];

    const int g = blockIdx.x, tid = threadIdx.x;
    const int wave = tid >> 5, lane = tid & 31, hi = lane >> 4, ln = lane & 15;

    // stage shared operands + per-graph features into LDS
    for (int idx = tid; idx < (64 * PITCH) / 8; idx += 256) {
        ((uint4*)sA2)[idx] = ((const uint4*)a2g)[idx];
        ((uint4*)sW2)[idx] = ((const uint4*)w2g)[idx];
    }
    for (int idx = tid; idx < N_NODES * IN_CH; idx += 256)
        sX[idx] = x[g * (N_NODES * IN_CH) + idx];
    for (int idx = tid; idx < HID * IN_CH; idx += 256) sW1[idx] = w1[idx];
    if (tid < 64) { sb1[tid] = b1[tid]; sb2[tid] = b2[tid]; sPool[tid] = 0.f; }
    __syncthreads();

    // T = X @ W1^T (K=5, fp32 VALU), stored transposed: Tt[c][r]
    for (int idx = tid; idx < 64 * 64; idx += 256) {
        const int c = idx >> 6, r = idx & 63;
        float v = 0.f;
        if (r < N_NODES) {
#pragma unroll
            for (int k = 0; k < IN_CH; ++k) v += sX[r * IN_CH + k] * sW1[c * IN_CH + k];
        }
        sB[c * PITCH + r] = (_Float16)v;
    }
    __syncthreads();

    // M1: H1 = relu(A2 @ T + b1), written transposed back over sB
    v8f acc[2] = {zero8(), zero8()};
    mm64(sA2, sB, wave, hi, ln, acc);
    __syncthreads();  // all Tt reads done before overwrite
#pragma unroll
    for (int lt = 0; lt < 2; ++lt) {
        const int t = wave * 2 + lt, tm = t >> 2, tn = t & 3;
        const int col = tn * 16 + ln;
#pragma unroll
        for (int i = 0; i < 8; ++i) {
            const int row = tm * 16 + hi * 8 + i;
            float h = acc[lt][i] + sb1[col];
            sB[col * PITCH + row] = (_Float16)(h > 0.f ? h : 0.f);
        }
    }
    __syncthreads();

    // M2: U = A2 @ H1 (no bias/relu yet), stored row-major
    acc[0] = zero8(); acc[1] = zero8();
    mm64(sA2, sB, wave, hi, ln, acc);
#pragma unroll
    for (int lt = 0; lt < 2; ++lt) {
        const int t = wave * 2 + lt, tm = t >> 2, tn = t & 3;
        const int col = tn * 16 + ln;
#pragma unroll
        for (int i = 0; i < 8; ++i)
            sU[(tm * 16 + hi * 8 + i) * PITCH + col] = (_Float16)acc[lt][i];
    }
    __syncthreads();

    // M3: H2 = relu(U @ W2^T + b2); fuse masked column-sum pooling
    acc[0] = zero8(); acc[1] = zero8();
    mm64(sU, sW2, wave, hi, ln, acc);
#pragma unroll
    for (int lt = 0; lt < 2; ++lt) {
        const int t = wave * 2 + lt, tm = t >> 2, tn = t & 3;
        const int col = tn * 16 + ln;
        float s = 0.f;
#pragma unroll
        for (int i = 0; i < 8; ++i) {
            const int row = tm * 16 + hi * 8 + i;
            if (row < N_NODES) {  // padded rows would contribute relu(b2)
                float h = acc[lt][i] + sb2[col];
                s += (h > 0.f ? h : 0.f);
            }
        }
        atomicAdd(&sPool[col], s);  // ds_add_f32
    }
    __syncthreads();

    // out[g] = pooled @ Wfc^T + bfc  (3x64 GEMV)
    if (tid < OUT_CH) {
        float o = bfc[tid];
        for (int n = 0; n < HID; ++n) o += sPool[n] * wfc[tid * HID + n];
        out[g * OUT_CH + tid] = o;
    }
}

extern "C" void kernel_launch(void* const* d_in, const int* in_sizes, int n_in,
                              void* d_out, int out_size, void* d_ws, size_t ws_size,
                              hipStream_t stream) {
    const float* x    = (const float*)d_in[0];
    // d_in[1] edge_index, d_in[2] batch: topology is static -> unused
    const float* tril = (const float*)d_in[3];
    const float* w1   = (const float*)d_in[4];
    const float* b1   = (const float*)d_in[5];
    const float* w2   = (const float*)d_in[6];
    const float* b2   = (const float*)d_in[7];
    const float* wfc  = (const float*)d_in[8];
    const float* bfc  = (const float*)d_in[9];

    _Float16* a2g = (_Float16*)d_ws;            // 64*72 f16 = 9216 B
    _Float16* w2g = a2g + 64 * PITCH;           // 64*72 f16 = 9216 B

    gcn_prep_kernel<<<2, 256, 0, stream>>>(tril, w2, a2g, w2g);
    gcn_graph_kernel<<<BATCH, 256, 0, stream>>>(x, w1, b1, b2, wfc, bfc,
                                                a2g, w2g, (float*)d_out);
}
